// Attention_27882927685773
// MI455X (gfx1250) — matrix operand order repaired and checked by reading, not compiled, not run on hardware
//
#include <hip/hip_runtime.h>
#include <cstdint>

typedef __attribute__((ext_vector_type(16))) _Float16 v16h;
typedef __attribute__((ext_vector_type(8)))  _Float16 v8h;
typedef __attribute__((ext_vector_type(8)))  float    v8f;
typedef __attribute__((ext_vector_type(4)))  int      v4i;

#define DIMS    2560
#define NHEADS  32
#define HD      80
#define HDP     96      // padded head dim (multiple of 32) in f16 buffers
#define SEQ     2048

// ---------------------------------------------------------------------------
// Async global->LDS staging (CDNA5 ASYNCcnt path) with sync fallback.
// Probe result: param0 = v4i addrspace(1)*, param1 = v4i addrspace(3)*.
// ---------------------------------------------------------------------------
#if __has_builtin(__builtin_amdgcn_global_load_async_to_lds_b128) && \
    __has_builtin(__builtin_amdgcn_s_wait_asynccnt)
#define USE_ASYNC_LDS 1
#else
#define USE_ASYNC_LDS 0
#endif

__device__ __forceinline__ void cp_async_b128(_Float16* lds_dst,
                                              const _Float16* gsrc) {
#if USE_ASYNC_LDS
  __builtin_amdgcn_global_load_async_to_lds_b128(
      (__attribute__((address_space(1))) v4i*)gsrc,
      (__attribute__((address_space(3))) v4i*)lds_dst,
      /*offset=*/0, /*cpol=*/0);
#else
  *(uint4*)lds_dst = *(const uint4*)gsrc;
#endif
}

__device__ __forceinline__ void cp_async_wait() {
#if USE_ASYNC_LDS
  __builtin_amdgcn_s_wait_asynccnt(0);
#endif
}

// ---------------------------------------------------------------------------
// WMMA helper: D = A(16x32 f16) * B(32x16 f16) + C(16x16 f32)
// ---------------------------------------------------------------------------
__device__ __forceinline__ v8f wmma16(v16h a, v16h b, v8f c) {
  return __builtin_amdgcn_wmma_f32_16x16x32_f16(
      false, a, false, b, (short)0, c, false, false);
}

// A-fragment gather (16x32 f16): lane row = lane&15, K-halves [2*kb..2*kb+15]
// (kb = (lane>>4)*8, i.e. K = (lane>>4)*16 + 0..15). p = row pointer, 4B aligned.
__device__ __forceinline__ v16h ld_afrag(const _Float16* p, int kb) {
  const unsigned int* q = (const unsigned int*)p;
  v16h r;
  unsigned int* rp = (unsigned int*)&r;
  const int d = kb;
#pragma unroll
  for (int i = 0; i < 8; ++i) rp[i] = q[d + i];
  return r;
}

// B-fragment gather (32x16 f16): lane col = lane&15, K = (lane>>4)*16 + 0..15,
// p = pointer to the 16 contiguous halves (4B aligned).
__device__ __forceinline__ v16h ld_bfrag(const _Float16* p) {
  const unsigned int* q = (const unsigned int*)p;
  v16h r;
  unsigned int* rp = (unsigned int*)&r;
#pragma unroll
  for (int i = 0; i < 8; ++i) rp[i] = q[i];
  return r;
}

// ---------------------------------------------------------------------------
// One-time fp32 -> f16 conversion (8 elements per thread).
// ---------------------------------------------------------------------------
__global__ __launch_bounds__(256)
void f32_to_f16(const float* __restrict__ src, _Float16* __restrict__ dst) {
  const size_t i = ((size_t)blockIdx.x * 256 + threadIdx.x) * 8;
  float4 a = *(const float4*)(src + i);
  float4 b = *(const float4*)(src + i + 4);
  v8h h;
  h[0] = (_Float16)a.x; h[1] = (_Float16)a.y;
  h[2] = (_Float16)a.z; h[3] = (_Float16)a.w;
  h[4] = (_Float16)b.x; h[5] = (_Float16)b.y;
  h[6] = (_Float16)b.z; h[7] = (_Float16)b.w;
  *(v8h*)(dst + i) = h;
}

// ---------------------------------------------------------------------------
// GEMM: C[m][n] = sum_k A[m][k]*B[n][k] + bias[n]  (A,B f16; accum f32)
// mode 0: C = float*, row-major [M][N].
// mode 1: N==3*DIMS; C = f16 base of q|k|v head-major [H][SEQ][HDP] (d<HD).
// Double-buffered async LDS staging, K-step 32.
// ---------------------------------------------------------------------------
__global__ __launch_bounds__(256)
void gemm_bias_f16(const _Float16* __restrict__ A, const _Float16* __restrict__ B,
                   const float* __restrict__ bias, void* __restrict__ Cv,
                   int M, int N, int K, int mode) {
  __shared__ _Float16 As[2 * 128 * 40];
  __shared__ _Float16 Bs[2 * 128 * 40];
  const int tid = threadIdx.x, wave = tid >> 5, lane = tid & 31;
  const int lo = lane & 15, hi = lane >> 4;
  const int m_tile = blockIdx.y * 128, n_tile = blockIdx.x * 128;
  const int wm = (wave >> 2) * 64;
  const int wn = (wave & 3) * 32;
  v8f acc[4][2] = {};

  auto stage = [&](int kt, int buf) {
    // 128 rows x 32 halfs = 512 x 16B chunks per matrix; 2 chunks/thread each.
#pragma unroll
    for (int u = 0; u < 2; ++u) {
      const int c = tid * 2 + u;
      const int row = c >> 2, seg = c & 3;
      cp_async_b128(As + buf * 5120 + row * 40 + seg * 8,
                    A + (size_t)(m_tile + row) * K + kt + seg * 8);
      cp_async_b128(Bs + buf * 5120 + row * 40 + seg * 8,
                    B + (size_t)(n_tile + row) * K + kt + seg * 8);
    }
  };

  stage(0, 0);
  cp_async_wait();
  __syncthreads();

  for (int kt = 0; kt < K; kt += 32) {
    const int cur = (kt >> 5) & 1;
    if (kt + 32 < K) stage(kt + 32, cur ^ 1);

    const _Float16* Ab = As + cur * 5120;
    const _Float16* Bb = Bs + cur * 5120;
    v16h af[4], bf[2];
#pragma unroll
    for (int mi = 0; mi < 4; ++mi)
      af[mi] = ld_afrag(Ab + (wm + mi * 16 + lo) * 40, hi * 8);
#pragma unroll
    for (int ni = 0; ni < 2; ++ni)
      bf[ni] = ld_bfrag(Bb + (wn + ni * 16 + lo) * 40 + hi * 16);
#pragma unroll
    for (int mi = 0; mi < 4; ++mi)
#pragma unroll
      for (int ni = 0; ni < 2; ++ni)
        acc[mi][ni] = wmma16(af[mi], bf[ni], acc[mi][ni]);

    cp_async_wait();
    __syncthreads();
  }

#pragma unroll
  for (int ni = 0; ni < 2; ++ni) {
    const int n = n_tile + wn + ni * 16 + lo;
    const float bv = bias[n];
    if (mode == 0) {
      float* C = (float*)Cv;
#pragma unroll
      for (int mi = 0; mi < 4; ++mi)
#pragma unroll
        for (int r = 0; r < 8; ++r) {
          const int m = m_tile + wm + mi * 16 + r + 8 * hi;
          C[(size_t)m * N + n] = acc[mi][ni][r] + bv;
        }
    } else {
      const int which = n / DIMS, rem = n % DIMS;
      const int h = rem / HD, d = rem % HD;
      _Float16* dst = (_Float16*)Cv +
                      (size_t)which * ((size_t)NHEADS * SEQ * HDP) +
                      ((size_t)h * SEQ) * HDP + d;
#pragma unroll
      for (int mi = 0; mi < 4; ++mi)
#pragma unroll
        for (int r = 0; r < 8; ++r) {
          const int m = m_tile + wm + mi * 16 + r + 8 * hi;
          dst[(size_t)m * HDP] = (_Float16)(acc[mi][ni][r] + bv);
        }
    }
  }
}

// ---------------------------------------------------------------------------
// In-place RoPE on f16 head-major q,k (rotate dims (d, d+16), d<16) and zero
// the pad columns [HD..HDP) of q, k, v.
// ---------------------------------------------------------------------------
__global__ __launch_bounds__(256)
void rope_pad(_Float16* __restrict__ qf, _Float16* __restrict__ kf,
              _Float16* __restrict__ vf) {
  const int idx = blockIdx.x * blockDim.x + threadIdx.x;
  const int h = idx / SEQ, l = idx % SEQ;
  if (h >= NHEADS) return;
  const size_t base = ((size_t)h * SEQ + l) * HDP;
#pragma unroll
  for (int part = 0; part < 2; ++part) {
    _Float16* p = (part ? kf : qf) + base;
#pragma unroll
    for (int d = 0; d < 16; ++d) {
      const float freq = __expf(-(float)d * 0.57564627324851148f); // ln(1e4)/16
      float sn, cs;
      __sincosf((float)l * freq, &sn, &cs);
      const float x1 = (float)p[d], x2 = (float)p[d + 16];
      p[d]      = (_Float16)(x1 * cs - x2 * sn);
      p[d + 16] = (_Float16)(x1 * sn + x2 * cs);
    }
#pragma unroll
    for (int d = HD; d < HDP; ++d) p[d] = (_Float16)0.f;
  }
#pragma unroll
  for (int d = HD; d < HDP; ++d) vf[base + d] = (_Float16)0.f;
}

// ---------------------------------------------------------------------------
// Flash attention (causal), f16 WMMA, double-buffered async K/V staging.
// One (head, 128 q-rows) per 256-thread block; each wave owns 16 q rows.
// Output written as f16 [SEQ][DIMS].
// ---------------------------------------------------------------------------
__global__ __launch_bounds__(256)
void flash_attn(const _Float16* __restrict__ Qh, const _Float16* __restrict__ Kh,
                const _Float16* __restrict__ Vh, _Float16* __restrict__ Oa) {
  constexpr int QS = 104;  // LDS row stride in halfs (96 data + pad)
  __shared__ _Float16 Qs[128 * QS];
  __shared__ _Float16 Ks[2 * 32 * QS];
  __shared__ _Float16 Vs[2 * 32 * QS];
  __shared__ _Float16 Ps[8 * 16 * 40];

  const int h = blockIdx.y;
  const int q0 = blockIdx.x * 128;
  const int tid = threadIdx.x, wave = tid >> 5, lane = tid & 31;
  const int lo = lane & 15, hi = lane >> 4;

  const _Float16* Qg = Qh + ((size_t)h * SEQ + q0) * HDP;
  const _Float16* Kg = Kh + (size_t)h * SEQ * HDP;
  const _Float16* Vg = Vh + (size_t)h * SEQ * HDP;

  auto stage_kv = [&](int kb, int buf) {
    const _Float16* Kb = Kg + (size_t)kb * 32 * HDP;
    const _Float16* Vb = Vg + (size_t)kb * 32 * HDP;
    _Float16* Kd = Ks + buf * (32 * QS);
    _Float16* Vd = Vs + buf * (32 * QS);
    for (int c = tid; c < 32 * 12; c += 256) {  // 12 x 16B chunks per row
      const int row = c / 12, seg = c % 12;
      cp_async_b128(Kd + row * QS + seg * 8, Kb + row * HDP + seg * 8);
      cp_async_b128(Vd + row * QS + seg * 8, Vb + row * HDP + seg * 8);
    }
  };

  // Stage the Q tile (128 rows x 96 halfs) and the first K/V block.
  for (int c = tid; c < 128 * 12; c += 256) {
    const int row = c / 12, seg = c % 12;
    cp_async_b128(Qs + row * QS + seg * 8, Qg + row * HDP + seg * 8);
  }
  stage_kv(0, 0);
  cp_async_wait();
  __syncthreads();

  v16h qa[3];
  {
    const _Float16* qrow = Qs + (wave * 16 + lo) * QS;
#pragma unroll
    for (int kk = 0; kk < 3; ++kk) qa[kk] = ld_afrag(qrow + kk * 32, hi * 8);
  }

  float mrow[8], lrow[8];
#pragma unroll
  for (int r = 0; r < 8; ++r) { mrow[r] = -1e30f; lrow[r] = 0.f; }
  v8f o[5] = {};

  const float scale = 0.11180339887498949f;  // 1/sqrt(80)
  const int qw = q0 + wave * 16;
  const int nkb = (q0 + 128) / 32;  // causal bound

  for (int kb = 0; kb < nkb; ++kb) {
    const int cur = kb & 1;
    if (kb + 1 < nkb) stage_kv(kb + 1, cur ^ 1);

    const int k_lo = kb * 32;
    if (k_lo <= qw + 15) {  // wave-uniform causal skip
      const _Float16* Kb = Ks + cur * (32 * QS);
      const _Float16* Vb = Vs + cur * (32 * QS);
      v8f s0 = {}, s1 = {};
#pragma unroll
      for (int kk = 0; kk < 3; ++kk) {
        v16h bk0 = ld_bfrag(Kb + lo * QS + kk * 32 + hi * 16);
        v16h bk1 = ld_bfrag(Kb + (16 + lo) * QS + kk * 32 + hi * 16);
        s0 = wmma16(qa[kk], bk0, s0);
        s1 = wmma16(qa[kk], bk1, s1);
      }
#pragma unroll
      for (int r = 0; r < 8; ++r) {
        const int qrow = qw + r + 8 * hi;
        float a0 = s0[r] * scale + (((k_lo + lo) > qrow) ? -1e9f : 0.f);
        float a1 = s1[r] * scale + (((k_lo + 16 + lo) > qrow) ? -1e9f : 0.f);
        float v = fmaxf(a0, a1);
#pragma unroll
        for (int off = 1; off < 16; off <<= 1)
          v = fmaxf(v, __shfl_xor(v, off, 32));
        const float mnew = fmaxf(mrow[r], v);
        const float alpha = __expf(mrow[r] - mnew);
        const float p0 = __expf(a0 - mnew);
        const float p1 = __expf(a1 - mnew);
        float ps = p0 + p1;
#pragma unroll
        for (int off = 1; off < 16; off <<= 1) ps += __shfl_xor(ps, off, 32);
        lrow[r] = lrow[r] * alpha + ps;
        mrow[r] = mnew;
#pragma unroll
        for (int t = 0; t < 5; ++t) o[t][r] = o[t][r] * alpha;
        Ps[(wave * 16 + r + 8 * hi) * 40 + lo]      = (_Float16)p0;
        Ps[(wave * 16 + r + 8 * hi) * 40 + 16 + lo] = (_Float16)p1;
      }
      v16h pa = ld_afrag(Ps + (wave * 16 + lo) * 40, hi * 8);
#pragma unroll
      for (int t = 0; t < 5; ++t) {
        v16h bv;
#pragma unroll
        for (int i = 0; i < 16; ++i)
          bv[i] = Vb[(hi * 16 + i) * QS + t * 16 + lo];
        o[t] = wmma16(pa, bv, o[t]);
      }
    }

    cp_async_wait();
    __syncthreads();
  }

#pragma unroll
  for (int t = 0; t < 5; ++t)
#pragma unroll
    for (int r = 0; r < 8; ++r) {
      const int qrow = qw + r + 8 * hi;
      Oa[(size_t)qrow * DIMS + h * HD + t * 16 + lo] =
          (_Float16)(o[t][r] / lrow[r]);
    }
}

// ---------------------------------------------------------------------------
// Launch: f32->f16 converts -> qkv gemm -> rope -> flash -> out-proj gemm
// ---------------------------------------------------------------------------
extern "C" void kernel_launch(void* const* d_in, const int* in_sizes, int n_in,
                              void* d_out, int out_size, void* d_ws, size_t ws_size,
                              hipStream_t stream) {
  (void)in_sizes; (void)n_in; (void)out_size; (void)ws_size;
  const float* x      = (const float*)d_in[0];  // [SEQ][DIMS]
  const float* Wqkv_w = (const float*)d_in[1];  // [3*DIMS][DIMS]
  const float* Wqkv_b = (const float*)d_in[2];  // [3*DIMS]
  const float* out_w  = (const float*)d_in[3];  // [DIMS][DIMS]
  const float* out_b  = (const float*)d_in[4];  // [DIMS]
  (void)d_in[5];                                // causal mask applied analytically

  _Float16* w16 = (_Float16*)d_ws;
  _Float16* xh  = w16;                                   // SEQ*DIMS
  _Float16* wh  = xh + (size_t)SEQ * DIMS;               // 3*DIMS*DIMS
  _Float16* owh = wh + (size_t)3 * DIMS * DIMS;          // DIMS*DIMS
  _Float16* qf  = owh + (size_t)DIMS * DIMS;             // [H][SEQ][HDP] x3
  _Float16* kf  = qf + (size_t)NHEADS * SEQ * HDP;
  _Float16* vf  = kf + (size_t)NHEADS * SEQ * HDP;
  _Float16* att = vf + (size_t)NHEADS * SEQ * HDP;       // SEQ*DIMS (f16)
  float* out = (float*)d_out;

  // 0) One-time f32 -> f16 conversions (2048 elems per block).
  f32_to_f16<<<(SEQ * DIMS) / 2048, 256, 0, stream>>>(x, xh);
  f32_to_f16<<<(3 * DIMS * DIMS) / 2048, 256, 0, stream>>>(Wqkv_w, wh);
  f32_to_f16<<<(DIMS * DIMS) / 2048, 256, 0, stream>>>(out_w, owh);

  // 1) QKV projection, written head-major f16 (96-padded rows).
  gemm_bias_f16<<<dim3(3 * DIMS / 128, SEQ / 128), 256, 0, stream>>>(
      xh, wh, Wqkv_b, qf, SEQ, 3 * DIMS, DIMS, /*mode=*/1);

  // 2) RoPE in place on q,k + zero pad columns.
  rope_pad<<<(SEQ * NHEADS) / 256, 256, 0, stream>>>(qf, kf, vf);

  // 3) Causal flash attention -> f16 [SEQ][DIMS].
  flash_attn<<<dim3(SEQ / 128, NHEADS), 256, 0, stream>>>(qf, kf, vf, att);

  // 4) Output projection -> f32 d_out.
  gemm_bias_f16<<<dim3(DIMS / 128, SEQ / 128), 256, 0, stream>>>(
      att, owh, out_b, out, SEQ, DIMS, DIMS, /*mode=*/0);
}
